// QuantizedBottleneck_62904091017996
// MI455X (gfx1250) — matrix operand order repaired and checked by reading, not compiled, not run on hardware
//
#include <hip/hip_runtime.h>

typedef __attribute__((ext_vector_type(16))) _Float16 v16h;
typedef __attribute__((ext_vector_type(8)))  float    v8f;

#define HW    196
#define BATCH 32
#define M_TOT (BATCH * HW)   // 6272 patches, divisible by 64

// ---------------------------------------------------------------------------
// Prep: lut [ncb][16][cout] f32  ->  lutT [ncb][cout][16] f16
// ---------------------------------------------------------------------------
__global__ void prep_lut_kernel(const float* __restrict__ lut,
                                _Float16* __restrict__ lutT,
                                int ncb, int cout) {
  int tid = blockIdx.x * blockDim.x + threadIdx.x;
  if (tid >= ncb * cout) return;
  int cb = tid / cout, o = tid - cb * cout;
  const float* src = lut + ((size_t)cb * 16) * cout + o;
  _Float16* dst = lutT + (size_t)tid * 16;
#pragma unroll
  for (int k = 0; k < 16; ++k) dst[k] = (_Float16)src[(size_t)k * cout];
}

// ---------------------------------------------------------------------------
// Assignment, layer 1: x NCHW [B,1024,14,14], dsub=4, ncb=256
// ---------------------------------------------------------------------------
__global__ void assign1_kernel(const float* __restrict__ x,
                               const float* __restrict__ cent,   // [256][16][4]
                               unsigned char* __restrict__ idx)  // [M][256]
{
  int m  = blockIdx.x * blockDim.x + threadIdx.x;
  int cb = blockIdx.y;
  if (m >= M_TOT) return;
  int b = m / HW, hw = m - b * HW;
  const float* xb = x + ((size_t)b * 1024 + 4 * cb) * HW + hw;
  float f0 = xb[0], f1 = xb[HW], f2 = xb[2 * HW], f3 = xb[3 * HW];
  const float* c = cent + (size_t)cb * 64;
  float best = 3.4e38f; int bk = 0;
#pragma unroll
  for (int k = 0; k < 16; ++k) {
    float c0 = c[4 * k], c1 = c[4 * k + 1], c2 = c[4 * k + 2], c3 = c[4 * k + 3];
    float d = c0 * (c0 - 2.f * f0) + c1 * (c1 - 2.f * f1)
            + c2 * (c2 - 2.f * f2) + c3 * (c3 - 2.f * f3);
    if (d < best) { best = d; bk = k; }
  }
  idx[(size_t)m * 256 + cb] = (unsigned char)bk;
}

// ---------------------------------------------------------------------------
// Assignment, layer 2: act1 [M][256] patch-major, 3x3 pad=1, dsub=9, ncb=256
// ---------------------------------------------------------------------------
__global__ void assign2_kernel(const float* __restrict__ act,   // [M][256]
                               const float* __restrict__ cent,  // [256][16][9]
                               unsigned char* __restrict__ idx) // [M][256]
{
  int m = blockIdx.x, c = threadIdx.x;
  int b = m / HW, p = m - b * HW, h = p / 14, w = p - h * 14;
  float f[9];
#pragma unroll
  for (int dh = -1; dh <= 1; ++dh)
#pragma unroll
    for (int dw = -1; dw <= 1; ++dw) {
      int hh = h + dh, ww = w + dw;
      int t = (dh + 1) * 3 + (dw + 1);
      f[t] = (hh >= 0 && hh < 14 && ww >= 0 && ww < 14)
               ? act[((size_t)b * HW + hh * 14 + ww) * 256 + c] : 0.f;
    }
  const float* cc = cent + (size_t)c * 144;
  float best = 3.4e38f; int bk = 0;
  for (int k = 0; k < 16; ++k) {
    float d = 0.f;
#pragma unroll
    for (int j = 0; j < 9; ++j) { float cv = cc[k * 9 + j]; d += cv * (cv - 2.f * f[j]); }
    if (d < best) { best = d; bk = k; }
  }
  idx[(size_t)m * 256 + c] = (unsigned char)bk;
}

// ---------------------------------------------------------------------------
// Assignment, layer 3: act2 [M][256] patch-major, dsub=4, ncb=64
// ---------------------------------------------------------------------------
__global__ void assign3_kernel(const float* __restrict__ act,   // [M][256]
                               const float* __restrict__ cent,  // [64][16][4]
                               unsigned char* __restrict__ idx) // [M][64]
{
  int m = blockIdx.x, cb = threadIdx.x;  // blockDim.x == 64
  const float4 f = *(const float4*)(act + (size_t)m * 256 + 4 * cb);
  const float* c = cent + (size_t)cb * 64;
  float best = 3.4e38f; int bk = 0;
#pragma unroll
  for (int k = 0; k < 16; ++k) {
    float c0 = c[4 * k], c1 = c[4 * k + 1], c2 = c[4 * k + 2], c3 = c[4 * k + 3];
    float d = c0 * (c0 - 2.f * f.x) + c1 * (c1 - 2.f * f.y)
            + c2 * (c2 - 2.f * f.z) + c3 * (c3 - 2.f * f.w);
    if (d < best) { best = d; bk = k; }
  }
  idx[(size_t)m * 64 + cb] = (unsigned char)bk;
}

// ---------------------------------------------------------------------------
// LUT accumulate as one-hot x LUT GEMM on WMMA f16.
//   - double-buffered LDS filled by GLOBAL_LOAD_ASYNC_TO_LDS_B128 (inline asm,
//     ASYNCcnt-tracked): no VGPR staging, no ds_store commit
//   - compute split into 4 sections of 4 tiles with hard sched_barrier walls
//     and ping-pong bA/bB fragment banks: loads for section g+1 are pinned a
//     full section ahead of their WMMAs, so the allocator cannot re-serialize
//   - per pair: issue async fill of buffer^1, 16 WMMAs, s_wait_asynccnt 0 +
//     barrier to publish
// Block = 128 threads (4 waves); each wave: one 16-row patch tile, NT 16x16
// f32 accumulator tiles (NT*16 output columns).
// ---------------------------------------------------------------------------
template<int NCB, int COUT, int NT, bool RELU, bool FINAL>
__global__ __launch_bounds__(128, 1)
void lut_accum_kernel(const unsigned char* __restrict__ idx, // [M][NCB]
                      const _Float16* __restrict__ lutT,     // [NCB][COUT][16]
                      const float* __restrict__ scale,
                      const float* __restrict__ bias,
                      const float* __restrict__ resid,       // x NCHW (FINAL only)
                      float* __restrict__ out)               // [M][COUT] or NCHW
{
  constexpr int NCOL = NT * 16;
  constexpr unsigned HVB  = NCOL * 16 * 2;      // bytes per codebook block (8KB)
  constexpr unsigned BUFB = 2 * HVB;            // bytes per LDS buffer (16KB)
  constexpr int NP = NCB / 2;                   // codebook pairs
  static_assert(NT == 16, "section pipeline assumes 16 tiles");
  __shared__ __align__(32) _Float16 lds[2 * 2 * NCOL * 16];  // two buffers

  const int lane = threadIdx.x & 31;
  const int wv   = threadIdx.x >> 5;
  const int m0   = (blockIdx.x * 4 + wv) * 16;
  const int col0 = blockIdx.y * NCOL;

  const int mrow = m0 + (lane & 15);
  const int koff = (lane < 16) ? 0 : 8;   // unused; A/B frags use contiguous k = 16*cbh + s
  const int cbh  = lane >> 4;             // which codebook of the pair (B frag)
  const unsigned char* idxrow = idx + (size_t)mrow * NCB;

  // LDS byte address for async-to-LDS vdst: low 32 bits of generic pointer
  const unsigned ldsbase = (unsigned)(size_t)(&lds[0]);

  v8f acc[NT] = {};
  unsigned short pi_cur, pi_next;

  auto issue_async = [&](int pair, int buf) {
    const char* g0 = (const char*)(lutT + ((size_t)(2 * pair)     * COUT + col0) * 16);
    const char* g1 = (const char*)(lutT + ((size_t)(2 * pair + 1) * COUT + col0) * 16);
    const unsigned lb   = ldsbase + (unsigned)buf * BUFB;
    const unsigned toff = (unsigned)threadIdx.x * 16u;
#pragma unroll
    for (int i = 0; i < (int)(HVB / 2048); ++i) {     // 128 thr x 16B = 2KB/op
      unsigned o = toff + (unsigned)i * 2048u;
      asm volatile("global_load_async_to_lds_b128 %0, %1, off"
                   :: "v"(lb + o), "v"(g0 + o) : "memory");
      asm volatile("global_load_async_to_lds_b128 %0, %1, off"
                   :: "v"(lb + HVB + o), "v"(g1 + o) : "memory");
    }
    pi_next = *(const unsigned short*)(idxrow + 2 * pair);
  };

  auto compute = [&](unsigned short pi, int buf) {
    const int ia = pi & 0xff, ib = pi >> 8;
    v16h a;
#pragma unroll
    for (int i = 0; i < 8; ++i) {
      a[i]     = ((cbh ? ib : ia) == i) ? (_Float16)1.0f : (_Float16)0.0f;
      a[i + 8] = ((cbh ? ib : ia) == i + 8) ? (_Float16)1.0f : (_Float16)0.0f;
    }
    const _Float16* lb = lds + (size_t)buf * (2 * NCOL * 16)
                       + ((size_t)cbh * NCOL + (lane & 15)) * 16;

    v16h bA[4], bB[4];
#pragma unroll
    for (int j = 0; j < 4; ++j) bA[j] = *(const v16h*)(lb + (size_t)j * 256);
    __builtin_amdgcn_sched_barrier(0);                    // wall: S0 done
#pragma unroll
    for (int g = 0; g < 4; ++g) {
      if ((g & 1) == 0) {
        if (g < 3) {
#pragma unroll
          for (int j = 0; j < 4; ++j)
            bB[j] = *(const v16h*)(lb + (size_t)((g + 1) * 4 + j) * 256);
        }
#pragma unroll
        for (int j = 0; j < 4; ++j)
          acc[g * 4 + j] = __builtin_amdgcn_wmma_f32_16x16x32_f16(
              false, a, false, bA[j], (short)0, acc[g * 4 + j], false, false);
      } else {
        if (g < 3) {
#pragma unroll
          for (int j = 0; j < 4; ++j)
            bA[j] = *(const v16h*)(lb + (size_t)((g + 1) * 4 + j) * 256);
        }
#pragma unroll
        for (int j = 0; j < 4; ++j)
          acc[g * 4 + j] = __builtin_amdgcn_wmma_f32_16x16x32_f16(
              false, a, false, bB[j], (short)0, acc[g * 4 + j], false, false);
      }
      __builtin_amdgcn_sched_group_barrier(0x100, 8, 0);  // loads first
      __builtin_amdgcn_sched_group_barrier(0x008, 4, 0);  // then 4 WMMAs
      __builtin_amdgcn_sched_barrier(0);                  // wall
    }
  };

  // Prologue: fill buffer 0
  issue_async(0, 0);
  pi_cur = pi_next;
  asm volatile("s_wait_asynccnt 0x0" ::: "memory");
  __syncthreads();

  for (int pair = 0; pair < NP - 1; ++pair) {
    issue_async(pair + 1, (pair + 1) & 1);  // async fill of the other buffer
    compute(pi_cur, pair & 1);              // 16 WMMAs from current buffer
    pi_cur = pi_next;
    asm volatile("s_wait_asynccnt 0x0" ::: "memory");  // my async writes done
    __syncthreads();                                   // everyone's done
  }

  compute(pi_cur, (NP - 1) & 1);            // peeled last pair

  // Epilogue: folded BN (+ReLU) / final residual add + ReLU + NCHW store
  const int ncol = lane & 15;
  const int mb   = m0 + ((lane >> 4) << 3);
#pragma unroll
  for (int t = 0; t < NT; ++t) {
    int n = col0 + t * 16 + ncol;
    float sc = scale[n], bi = bias[n];
#pragma unroll
    for (int r = 0; r < 8; ++r) {
      int m = mb + r;
      float v = acc[t][r] * sc + bi;
      if (FINAL) {
        int b_ = m / HW, p = m - b_ * HW;
        size_t oi = ((size_t)b_ * COUT + n) * HW + p;
        v += resid[oi];
        out[oi] = fmaxf(v, 0.f);
      } else {
        if (RELU) v = fmaxf(v, 0.f);
        out[(size_t)m * COUT + n] = v;
      }
    }
  }
}

// ---------------------------------------------------------------------------
extern "C" void kernel_launch(void* const* d_in, const int* in_sizes, int n_in,
                              void* d_out, int out_size, void* d_ws, size_t ws_size,
                              hipStream_t stream) {
  (void)in_sizes; (void)n_in; (void)out_size; (void)ws_size;

  const float* x   = (const float*)d_in[0];
  const float* c1c = (const float*)d_in[1];
  const float* c1l = (const float*)d_in[2];
  const float* c1s = (const float*)d_in[3];
  const float* c1b = (const float*)d_in[4];
  const float* c2c = (const float*)d_in[5];
  const float* c2l = (const float*)d_in[6];
  const float* c2s = (const float*)d_in[7];
  const float* c2b = (const float*)d_in[8];
  const float* c3c = (const float*)d_in[9];
  const float* c3l = (const float*)d_in[10];
  const float* c3s = (const float*)d_in[11];
  const float* c3b = (const float*)d_in[12];
  float* out = (float*)d_out;

  // Workspace layout (all sizes are multiples of 256B): ~21.7 MB total
  char* ws = (char*)d_ws;
  constexpr size_t SZ_LUT12 = (size_t)256 * 256 * 16 * 2;   // 2 MB each
  constexpr size_t SZ_LUT3  = (size_t)64 * 1024 * 16 * 2;   // 2 MB
  constexpr size_t SZ_IDX12 = (size_t)M_TOT * 256;          // 1.6 MB each
  constexpr size_t SZ_IDX3  = (size_t)M_TOT * 64;           // 0.4 MB
  constexpr size_t SZ_ACT   = (size_t)M_TOT * 256 * 4;      // 6.4 MB each

  _Float16* lut1 = (_Float16*)(ws);
  _Float16* lut2 = (_Float16*)(ws + SZ_LUT12);
  _Float16* lut3 = (_Float16*)(ws + 2 * SZ_LUT12);
  unsigned char* idx1 = (unsigned char*)(ws + 2 * SZ_LUT12 + SZ_LUT3);
  unsigned char* idx2 = idx1 + SZ_IDX12;
  unsigned char* idx3 = idx2 + SZ_IDX12;
  float* act1 = (float*)((char*)idx3 + SZ_IDX3);
  float* act2 = (float*)((char*)act1 + SZ_ACT);

  // LUT f32 -> f16 transposed [cb][o][k]
  prep_lut_kernel<<<dim3((256 * 256 + 255) / 256), 256, 0, stream>>>(c1l, lut1, 256, 256);
  prep_lut_kernel<<<dim3((256 * 256 + 255) / 256), 256, 0, stream>>>(c2l, lut2, 256, 256);
  prep_lut_kernel<<<dim3((64 * 1024 + 255) / 256), 256, 0, stream>>>(c3l, lut3, 64, 1024);

  // Layer 1: 1x1 AMM conv + BN + ReLU
  assign1_kernel<<<dim3((M_TOT + 255) / 256, 256), 256, 0, stream>>>(x, c1c, idx1);
  lut_accum_kernel<256, 256, 16, true, false>
      <<<dim3(M_TOT / 64, 1), 128, 0, stream>>>(idx1, lut1, c1s, c1b, nullptr, act1);

  // Layer 2: 3x3 AMM conv + BN + ReLU
  assign2_kernel<<<dim3(M_TOT), 256, 0, stream>>>(act1, c2c, idx2);
  lut_accum_kernel<256, 256, 16, true, false>
      <<<dim3(M_TOT / 64, 1), 128, 0, stream>>>(idx2, lut2, c2s, c2b, nullptr, act2);

  // Layer 3: 1x1 AMM conv + BN, fused residual add + ReLU, NCHW output
  assign3_kernel<<<dim3(M_TOT), 64, 0, stream>>>(act2, c3c, idx3);
  lut_accum_kernel<64, 1024, 16, false, true>
      <<<dim3(M_TOT / 64, 4), 128, 0, stream>>>(idx3, lut3, c3s, c3b, x, out);
}
